// SeqSelfAttention_14499809591629
// MI455X (gfx1250) — compile-verified, matrix operand order checked by reading, not run on hardware
//
#include <hip/hip_runtime.h>
#include <hip/hip_bf16.h>
#include <math.h>

#define BATCH 8
#define SLEN  2048
#define DIM   512

typedef __attribute__((ext_vector_type(16))) __bf16 v16bf;
typedef __attribute__((ext_vector_type(8)))  __bf16 v8bf;
typedef __attribute__((ext_vector_type(8)))  float  v8f;
typedef __attribute__((ext_vector_type(4)))  float  v4f;
typedef __attribute__((ext_vector_type(4)))  int    v4i;

// ---------------------------------------------------------------------------
// CDNA5 async global->LDS copy (ASYNCcnt path), with sync fallback
// ---------------------------------------------------------------------------
#if defined(__has_builtin)
#if __has_builtin(__builtin_amdgcn_global_load_async_to_lds_b128)
#define HAVE_ASYNC_LDS 1
#endif
#endif
#ifndef HAVE_ASYNC_LDS
#define HAVE_ASYNC_LDS 0
#endif

__device__ __forceinline__ void async_copy16(const __bf16* g, __bf16* l) {
#if HAVE_ASYNC_LDS
  __builtin_amdgcn_global_load_async_to_lds_b128((v4i*)g, (v4i*)l,
                                                 /*offset=*/0, /*cpol=*/0);
#else
  *(v8bf*)l = *(const v8bf*)g;
#endif
}

__device__ __forceinline__ void async_wait0() {
#if HAVE_ASYNC_LDS
  asm volatile("s_wait_asynccnt 0" ::: "memory");
#endif
}

// ---------------------------------------------------------------------------
// WMMA helper: D = A(16x32 bf16) x B(32x16 bf16) + C(16x16 f32)
// ---------------------------------------------------------------------------
__device__ __forceinline__ v8f wmma_bf16(v16bf a, v16bf b, v8f c) {
  return __builtin_amdgcn_wmma_f32_16x16x32_bf16(
      /*neg_a=*/false, a, /*neg_b=*/false, b,
      /*c_mod=*/(short)0, c, /*reuse_a=*/false, /*reuse_b=*/false);
}

__device__ __forceinline__ v8f zero8() {
  v8f z;
#pragma unroll
  for (int i = 0; i < 8; ++i) z[i] = 0.0f;
  return z;
}

// ---------------------------------------------------------------------------
// Load one 16-bit WMMA operand tile (A: 16 rows x 32 k  /  B: 32 k x 16 cols
// where each row/col is contiguous in memory with leading dim `ld`).
// Per ISA 7.12.2: lanes 0-15 hold row/col (lane&15) with k=0..7 (VGPR0-3) and
// k=16..23 (VGPR4-7); lanes 16-31 hold the same row with k+8. Two b128 loads.
// ---------------------------------------------------------------------------
__device__ __forceinline__ v16bf load_tile16(const __bf16* base, int ld,
                                             int k0, int lane) {
  int rc = lane & 15;
  int hi = lane >> 4;
  const __bf16* p = base + (size_t)rc * ld + k0 + hi * 8;
  v8bf c0 = *(const v8bf*)(p);
  v8bf c1 = *(const v8bf*)(p + 16);
  v16bf r;
#pragma unroll
  for (int i = 0; i < 8; ++i) { r[i] = c0[i]; r[i + 8] = c1[i]; }
  return r;
}

// ---------------------------------------------------------------------------
// Kernel 1: f32 -> bf16 bulk convert (X -> Xb), 8 elements / thread
// ---------------------------------------------------------------------------
__global__ void cvt_bf16_kernel(const float* __restrict__ x,
                                __bf16* __restrict__ xb, int n) {
  int i = (blockIdx.x * blockDim.x + threadIdx.x) * 8;
  if (i >= n) return;
  v4f a = *(const v4f*)(x + i);
  v4f b = *(const v4f*)(x + i + 4);
  v8bf o;
#pragma unroll
  for (int j = 0; j < 4; ++j) { o[j] = (__bf16)a[j]; o[j + 4] = (__bf16)b[j]; }
  *(v8bf*)(xb + i) = o;
}

// ---------------------------------------------------------------------------
// Kernel 2: Wt[e][d] = bf16(W[d][e])  (columns of W become contiguous rows)
// ---------------------------------------------------------------------------
__global__ void cvt_wt_kernel(const float* __restrict__ w,
                              __bf16* __restrict__ wt) {
  int idx = blockIdx.x * blockDim.x + threadIdx.x;   // DIM*DIM threads
  int e = idx >> 9, d = idx & (DIM - 1);
  wt[idx] = (__bf16)w[d * DIM + e];
}

// ---------------------------------------------------------------------------
// Kernel 3: tiled transpose Xb[b][s][d] -> Xt[b][d][s]  (V^T, bf16)
// ---------------------------------------------------------------------------
__global__ void transpose_xt_kernel(const __bf16* __restrict__ xb,
                                    __bf16* __restrict__ xt) {
  __shared__ __bf16 tile[32][33];
  int b = blockIdx.z;
  int s0 = blockIdx.x * 32, d0 = blockIdx.y * 32;
  int tx = threadIdx.x, ty = threadIdx.y;            // 32 x 8
#pragma unroll
  for (int i = 0; i < 4; ++i) {
    int s = ty + i * 8;
    tile[s][tx] = xb[((size_t)(b * SLEN + s0 + s)) * DIM + d0 + tx];
  }
  __syncthreads();
#pragma unroll
  for (int i = 0; i < 4; ++i) {
    int d = ty + i * 8;
    xt[((size_t)(b * DIM + d0 + d)) * SLEN + s0 + tx] = tile[tx][d];
  }
}

// ---------------------------------------------------------------------------
// Kernel 4: Q = X @ W  (bf16 WMMA, wave -> 16x64 tile, software pipelined)
// ---------------------------------------------------------------------------
__global__ __launch_bounds__(256) void proj_kernel(
    const __bf16* __restrict__ xb, const __bf16* __restrict__ wt,
    __bf16* __restrict__ qb) {
  int lane = threadIdx.x & 31;
  int wid  = (blockIdx.x * blockDim.x + threadIdx.x) >> 5;
  int nt = wid & 7;            // 64-wide n slab
  int mt = (wid >> 3) & 127;   // 16-row m tile
  int b  = wid >> 10;

  const __bf16* abase = xb + ((size_t)(b * SLEN + mt * 16)) * DIM;
  const __bf16* wbase = wt + (size_t)(nt * 64) * DIM;

  v8f acc[4];
#pragma unroll
  for (int j = 0; j < 4; ++j) acc[j] = zero8();

  v16bf a = load_tile16(abase, DIM, 0, lane);
  v16bf bt[4];
#pragma unroll
  for (int j = 0; j < 4; ++j)
    bt[j] = load_tile16(wbase + (size_t)(j * 16) * DIM, DIM, 0, lane);

#pragma unroll
  for (int k0 = 0; k0 < DIM - 32; k0 += 32) {
    v16bf an = load_tile16(abase, DIM, k0 + 32, lane);
    v16bf bn[4];
#pragma unroll
    for (int j = 0; j < 4; ++j)
      bn[j] = load_tile16(wbase + (size_t)(j * 16) * DIM, DIM, k0 + 32, lane);
#pragma unroll
    for (int j = 0; j < 4; ++j) acc[j] = wmma_bf16(a, bt[j], acc[j]);
    a = an;
#pragma unroll
    for (int j = 0; j < 4; ++j) bt[j] = bn[j];
  }
#pragma unroll
  for (int j = 0; j < 4; ++j) acc[j] = wmma_bf16(a, bt[j], acc[j]);

  int col = lane & 15, hi = lane >> 4;
#pragma unroll
  for (int j = 0; j < 4; ++j)
#pragma unroll
    for (int r = 0; r < 8; ++r) {
      int m = hi * 8 + r;
      qb[((size_t)(b * SLEN + mt * 16 + m)) * DIM + nt * 64 + j * 16 + col] =
          (__bf16)acc[j][r];
    }
}

// ---------------------------------------------------------------------------
// Kernel 5: fused attention.  One block (8 waves) per (b, 16-row query tile).
//   K rows are async-staged to LDS one chunk ahead (wave-local region, no
//   barrier needed -> stage 2 is pure LDS x LDS WMMA).  tanh bounds scores so
//   softmax needs no online max.  Wave w: 16 keys (stage 2), then 64-wide
//   O^T d-slice via O^T = V^T P^T (stage 3, pipelined global loads).
// ---------------------------------------------------------------------------
#define QSTRIDE 520   // 512 + 8 pad : conflict-free b128 LDS rows
#define PSTRIDE 136   // 128 + 8 pad

// wave w stages its 16 K rows of chunk kv0 into ldsK rows [16w, 16w+16)
__device__ __forceinline__ void stage_K(const __bf16* kb_b, __bf16* ldsK,
                                        int kv0, int w, int lane) {
  const __bf16* g0 = kb_b + (size_t)(kv0 + 16 * w) * DIM;
  __bf16* l0 = ldsK + (16 * w) * QSTRIDE;
#pragma unroll
  for (int i = 0; i < 32; ++i) {           // 32 x (32 lanes x 16B) = 16KB
    int row = i >> 1, half = i & 1;
    async_copy16(g0 + (size_t)row * DIM + half * 256 + lane * 8,
                 l0 + row * QSTRIDE + half * 256 + lane * 8);
  }
}

__global__ __launch_bounds__(256) void attn_kernel(
    const __bf16* __restrict__ xb,   // K  [B][S][D]
    const __bf16* __restrict__ xt,   // V^T [B][D][S]
    const __bf16* __restrict__ qb,   // Q  [B][S][D]
    float* __restrict__ out) {       // O  [B][S][D] f32
  __shared__ __bf16 ldsQ[16 * QSTRIDE];
  __shared__ __bf16 ldsK[128 * QSTRIDE];   // 130 KB, chunk of 128 key rows
  __shared__ __bf16 ldsP[16 * PSTRIDE];
  __shared__ float  ldsL[8][16];

  int tid  = threadIdx.x;
  int lane = tid & 31;
  int w    = tid >> 5;               // wave 0..7
  int col  = lane & 15, hi = lane >> 4;

  int m0 = (blockIdx.x & 127) * 16;
  int b  = blockIdx.x >> 7;

  const __bf16* kb_b   = xb + (size_t)b * SLEN * DIM;
  const __bf16* vtbase = xt + ((size_t)(b * DIM + w * 64)) * SLEN;

  // ---- stage 0: Q tile -> LDS; async-stage first K chunk ------------------
  {
    int m = tid >> 4, seg = tid & 15;
    const __bf16* src = qb + ((size_t)(b * SLEN + m0 + m)) * DIM + seg * 32;
    __bf16* dst = ldsQ + m * QSTRIDE + seg * 32;
#pragma unroll
    for (int i = 0; i < 4; ++i)
      *(v8bf*)(dst + i * 8) = *(const v8bf*)(src + i * 8);
  }
  stage_K(kb_b, ldsK, 0, w, lane);
  __syncthreads();

  v8f oacc[4];
#pragma unroll
  for (int j = 0; j < 4; ++j) oacc[j] = zero8();
  float lsum[8];
#pragma unroll
  for (int r = 0; r < 8; ++r) lsum[r] = 0.0f;

  const __bf16* kl = ldsK + (16 * w) * QSTRIDE;   // this wave's key rows

  for (int kv0 = 0; kv0 < SLEN; kv0 += 128) {
    // warm next chunk's V^T slab (64 rows x 256B = 128 cachelines)
    if (kv0 + 128 < SLEN) {
#pragma unroll
      for (int i = 0; i < 4; ++i) {
        int row = lane * 2 + (i >> 1);
        __builtin_prefetch(vtbase + (size_t)row * SLEN + kv0 + 128 +
                               (i & 1) * 64, 0, 3);
      }
    }

    // ---- stage 2: S = Q K^T, all operands in LDS -------------------------
    async_wait0();                      // K chunk ready (wave-local region)
    v8f s = zero8();
#pragma unroll
    for (int k0 = 0; k0 < DIM; k0 += 32) {
      v16bf a  = load_tile16(ldsQ, QSTRIDE, k0, lane);
      v16bf bm = load_tile16(kl, QSTRIDE, k0, lane);
      s = wmma_bf16(a, bm, s);
    }

    // p = exp(tanh(s)) in [e^-1, e]: no max-subtraction needed
#pragma unroll
    for (int r = 0; r < 8; ++r) {
      float p = __expf(tanhf(s[r]));
      lsum[r] += p;                                     // this lane's column
      ldsP[(hi * 8 + r) * PSTRIDE + 16 * w + col] = (__bf16)p;
    }
    __syncthreads();

    // issue async staging of NEXT K chunk; overlaps all of stage 3
    if (kv0 + 128 < SLEN) stage_K(kb_b, ldsK, kv0 + 128, w, lane);

    // ---- stage 3: O^T[64w..64w+63][m] += V^T P^T (pipelined, 16 tiles) ---
    {
      v16bf av = load_tile16(vtbase, SLEN, kv0, lane);
      v16bf bp = load_tile16(ldsP, PSTRIDE, 0, lane);
#pragma unroll
      for (int t = 0; t < 15; ++t) {
        int tn = t + 1;
        v16bf an = load_tile16(vtbase + (size_t)((tn >> 2) * 16) * SLEN, SLEN,
                               kv0 + (tn & 3) * 32, lane);
        v16bf bn = load_tile16(ldsP, PSTRIDE, (tn & 3) * 32, lane);
        oacc[t >> 2] = wmma_bf16(av, bp, oacc[t >> 2]);
        av = an; bp = bn;
      }
      oacc[3] = wmma_bf16(av, bp, oacc[3]);
    }
    __syncthreads();
  }

  // ---- row-sum reduction: 16 lanes per half, then 8 waves via LDS ---------
#pragma unroll
  for (int r = 0; r < 8; ++r) {
    float v = lsum[r];
#pragma unroll
    for (int m = 1; m < 16; m <<= 1) v += __shfl_xor(v, m, 32);
    lsum[r] = v;
  }
  if (col == 0) {
#pragma unroll
    for (int r = 0; r < 8; ++r) ldsL[w][hi * 8 + r] = lsum[r];
  }
  __syncthreads();

  float L = 0.0f;
#pragma unroll
  for (int ww = 0; ww < 8; ++ww) L += ldsL[ww][col];
  float rinv = 1.0f / L;   // this lane's 8 O^T values all share row m0+col

  // ---- store O (coalesced b128 f32 stores) --------------------------------
  float* obase = out + ((size_t)(b * SLEN + m0 + col)) * DIM + w * 64 + hi * 8;
#pragma unroll
  for (int dt = 0; dt < 4; ++dt) {
    v4f t0, t1;
#pragma unroll
    for (int r = 0; r < 4; ++r) { t0[r] = oacc[dt][r] * rinv; t1[r] = oacc[dt][r + 4] * rinv; }
    *(v4f*)(obase + dt * 16)     = t0;
    *(v4f*)(obase + dt * 16 + 4) = t1;
  }
}

// ---------------------------------------------------------------------------
extern "C" void kernel_launch(void* const* d_in, const int* in_sizes, int n_in,
                              void* d_out, int out_size, void* d_ws,
                              size_t ws_size, hipStream_t stream) {
  (void)in_sizes; (void)n_in; (void)out_size; (void)ws_size;
  const float* x   = (const float*)d_in[0];   // [B,S,D] f32
  const float* wgt = (const float*)d_in[1];   // [D,D]  f32
  float* out = (float*)d_out;                 // [B,S,D] f32

  char* ws = (char*)d_ws;
  __bf16* Xb = (__bf16*)(ws);                        // 16 MB
  __bf16* Qb = (__bf16*)(ws + ((size_t)16 << 20));   // 16 MB
  __bf16* Xt = (__bf16*)(ws + ((size_t)32 << 20));   // 16 MB
  __bf16* Wt = (__bf16*)(ws + ((size_t)48 << 20));   // 512 KB

  const int nX = BATCH * SLEN * DIM;                 // 8,388,608

  cvt_bf16_kernel<<<nX / (256 * 8), 256, 0, stream>>>(x, Xb, nX);
  cvt_wt_kernel<<<(DIM * DIM) / 256, 256, 0, stream>>>(wgt, Wt);
  transpose_xt_kernel<<<dim3(SLEN / 32, DIM / 32, BATCH), dim3(32, 8), 0,
                        stream>>>(Xb, Xt);
  proj_kernel<<<(BATCH * (SLEN / 16) * (DIM / 64)) / 8, 256, 0, stream>>>(
      Xb, Wt, Qb);
  attn_kernel<<<BATCH * (SLEN / 16), 256, 0, stream>>>(Xb, Xt, Qb, out);
}
